// Physics_Attention_Structured_Mesh_2D_Auto_Encoder_61349312856327
// MI455X (gfx1250) — matrix operand order repaired and checked by reading, not compiled, not run on hardware
//
#include <hip/hip_runtime.h>
#include <stdint.h>

// ---------------------------------------------------------------------------
// Problem constants (reference: H=W=256, HEADS=8, DIM_HEAD=64, SLICE=64, C=128)
// ---------------------------------------------------------------------------
#define H_    256
#define W_    256
#define NTOK  65536        // H*W
#define CIN   128
#define HEADS 8
#define DH    64
#define INNER 512          // HEADS*DH
#define OC2   1024         // both convs fused: [0,512)=fx, [512,1024)=x
#define KTOT  1152         // 9 taps * 128 channels
#define GSL   64           // SLICE
#define BATCH 2

typedef __attribute__((ext_vector_type(16))) __bf16 v16bf;
typedef __attribute__((ext_vector_type(8)))  float  v8f;

union BF2 { uint32_t u; __bf16 h[2]; };

__device__ __forceinline__ v8f v8f_zero() {
  v8f r;
#pragma unroll
  for (int i = 0; i < 8; ++i) r[i] = 0.0f;
  return r;
}

// A-matrix (16x32 bf16) element->K mapping. e = packed element index 0..15,
// half = lane/16:  k = half*16 + e  (contiguous, matches B fragments)
__device__ __forceinline__ int a_kofs(int e, int half) {
  return half * 16 + e;
}
// B-matrix (32x16 bf16): lanes 0-15 hold K 0..15 (k=e), lanes 16-31 K 16..31.

__device__ __forceinline__ v8f wmma_bf16(v16bf a, v16bf b, v8f c) {
  // (neg_a, A, neg_b, B, c_mod, C, reuse_a, reuse_b)
  return __builtin_amdgcn_wmma_f32_16x16x32_bf16(false, a, false, b,
                                                 (short)0, c, false, false);
}

// ---------------------------------------------------------------------------
// Prep kernels: f32 -> bf16 conversions + zero-init of accumulators
// ---------------------------------------------------------------------------
__global__ void prep_weights(const float* __restrict__ Wfx,
                             const float* __restrict__ Wx,
                             __bf16* __restrict__ wT) {
  int idx = blockIdx.x * blockDim.x + threadIdx.x;
  if (idx >= OC2 * KTOT) return;
  int oc = idx / KTOT, k = idx - oc * KTOT;
  const float* src = (oc < INNER) ? Wfx : Wx;      // HWIO: [k][oc%512]
  wT[(size_t)oc * KTOT + k] = (__bf16)src[(size_t)k * INNER + (oc & (INNER - 1))];
}

__global__ void prep_x(const float* __restrict__ x, __bf16* __restrict__ xbf, int n) {
  int idx = blockIdx.x * blockDim.x + threadIdx.x;
  if (idx < n) xbf[idx] = (__bf16)x[idx];
}

__global__ void prep_misc(const float* __restrict__ Wslice, __bf16* __restrict__ wsl,
                          float* __restrict__ norm, float* __restrict__ acc) {
  int idx = blockIdx.x * blockDim.x + threadIdx.x;
  if (idx < GSL * DH)                 wsl[idx] = (__bf16)Wslice[idx];
  if (idx < BATCH * HEADS * GSL)      norm[idx] = 0.0f;
  if (idx < BATCH * HEADS * GSL * DH) acc[idx] = 0.0f;
}

// ---------------------------------------------------------------------------
// Kernel A: dual 3x3 conv as implicit GEMM (bf16 WMMA, f32 accum)
// block = 64 pixels (one row segment) x 256 combined output channels.
// 8 waves: wave w owns oc columns [w*32, w*32+32) and ALL 4 pixel tiles
// (4 M-tiles x 2 N-tiles = 8 accumulators) so each B-fragment pair feeds
// 8 WMMAs -> 4x less L2 weight traffic than a 1x2 blocking.
// ---------------------------------------------------------------------------
__global__ __launch_bounds__(256) void conv_dual(
    const __bf16* __restrict__ xbf, const __bf16* __restrict__ wT,
    const float* __restrict__ bfx, const float* __restrict__ bx,
    __bf16* __restrict__ mid) {
  extern __shared__ char smem_raw[];
  __bf16*   ldsIn  = (__bf16*)smem_raw;       // [3*66][128] halo tile, bf16
  uint32_t* ldsInU = (uint32_t*)smem_raw;

  int blk = blockIdx.x;
  int ocQuad = blk & 3;   blk >>= 2;   // which 256-wide oc slab
  int tileM  = blk & 1023; blk >>= 10;
  int b = blk;
  int y  = tileM >> 2;
  int x0 = (tileM & 3) << 6;
  int n0 = y * W_ + x0;

  int tid = threadIdx.x;
  // ---- stage input halo: 3 rows x 66 cols x 128 ch (zero-padded SAME) ----
  if (tid < 198) {
    int pr = tid / 66, pc = tid - pr * 66;
    int gy = y + pr - 1, gx = x0 + pc - 1;
    uint32_t* dst = ldsInU + tid * (CIN / 2);
    if (gy >= 0 && gy < H_ && gx >= 0 && gx < W_) {
      const uint32_t* src =
          (const uint32_t*)(xbf + (size_t)(b * NTOK + gy * W_ + gx) * CIN);
#pragma unroll 4
      for (int i = 0; i < CIN / 2; ++i) dst[i] = src[i];
    } else {
#pragma unroll 4
      for (int i = 0; i < CIN / 2; ++i) dst[i] = 0u;
    }
  }
  __syncthreads();

  int w = tid >> 5, lane = tid & 31;
  int half = lane >> 4, lo = lane & 15;
  int ocg = ocQuad * 256 + w * 32;
  int oc0 = ocg + lo;
  int oc1 = ocg + 16 + lo;

  const __bf16* wrow0 = wT + (size_t)oc0 * KTOT + half * 16;
  const __bf16* wrow1 = wT + (size_t)oc1 * KTOT + half * 16;

  v8f acc[8];   // [m_tile][n_tile] -> acc[m*2+t]
#pragma unroll
  for (int i = 0; i < 8; ++i) acc[i] = v8f_zero();

  for (int dy = 0; dy < 3; ++dy) {
    for (int dx = 0; dx < 3; ++dx) {
      int tap = dy * 3 + dx;
      const __bf16* abase = ldsIn + (dy * 66 + lo + dx) * CIN;
      __builtin_prefetch(wrow0 + tap * 128 + 128, 0, 1);
      __builtin_prefetch(wrow1 + tap * 128 + 128, 0, 1);
#pragma unroll
      for (int kc = 0; kc < 4; ++kc) {
        int kb = tap * 128 + kc * 32;
        v16bf b0, b1;
#pragma unroll
        for (int e = 0; e < 16; ++e) b0[e] = wrow0[kb + e];
#pragma unroll
        for (int e = 0; e < 16; ++e) b1[e] = wrow1[kb + e];
#pragma unroll
        for (int m = 0; m < 4; ++m) {
          const __bf16* arow = abase + (m * 16) * CIN;   // pixel m*16+lo
          v16bf a;
#pragma unroll
          for (int e = 0; e < 16; ++e) a[e] = arow[kc * 32 + a_kofs(e, half)];
          acc[m * 2]     = wmma_bf16(a, b0, acc[m * 2]);
          acc[m * 2 + 1] = wmma_bf16(a, b1, acc[m * 2 + 1]);
        }
      }
    }
  }

  float bias0 = (oc0 < INNER) ? bfx[oc0] : bx[oc0 - INNER];
  float bias1 = (oc1 < INNER) ? bfx[oc1] : bx[oc1 - INNER];
#pragma unroll
  for (int m = 0; m < 4; ++m) {
#pragma unroll
    for (int v = 0; v < 8; ++v) {
      int prow = m * 16 + v + 8 * half;
      size_t base = (size_t)(b * NTOK + n0 + prow) * OC2;
      mid[base + oc0] = (__bf16)(acc[m * 2][v] + bias0);
      mid[base + oc1] = (__bf16)(acc[m * 2 + 1][v] + bias1);
    }
  }
}

// ---------------------------------------------------------------------------
// Kernel B: per (b,h,64-token tile): logits WMMA -> softmax -> store slice
// weights (g-major), atomic slice_norm, WMMA slice_token partial + atomics.
// ---------------------------------------------------------------------------
__global__ __launch_bounds__(256) void slice_kernel(
    const __bf16* __restrict__ mid, const __bf16* __restrict__ wsl,
    const float* __restrict__ bslice, const float* __restrict__ temperature,
    __bf16* __restrict__ swg_all, float* __restrict__ norm,
    float* __restrict__ acc) {
  extern __shared__ char smem_raw[];
  __bf16* lds_xm  = (__bf16*)smem_raw;          // [64 tok][64 d]
  __bf16* lds_fx  = lds_xm + 64 * 64;           // [64 tok][64 d]
  __bf16* lds_swT = lds_fx + 64 * 64;           // [64 g][64 tok]
  float*  lds_lg  = (float*)(lds_swT + 64 * 64);// [64 tok][65] (padded)

  int blk  = blockIdx.x;
  int tile = blk & 1023;
  int bh   = blk >> 10;
  int hh   = bh & 7;
  int b    = bh >> 3;
  int n0   = tile * 64;
  int tid  = threadIdx.x;

  { // stage x_mid / fx tiles (channels h*64..h*64+63 of mid)
    int token = tid >> 2;
    int c0 = (tid & 3) * 16;
    const uint32_t* srcRow =
        (const uint32_t*)(mid + (size_t)(b * NTOK + n0 + token) * OC2);
    uint32_t* dx = (uint32_t*)(lds_xm + token * 64 + c0);
    uint32_t* df = (uint32_t*)(lds_fx + token * 64 + c0);
    int xoff = (INNER + hh * DH + c0) >> 1;
    int foff = (hh * DH + c0) >> 1;
#pragma unroll
    for (int i = 0; i < 8; ++i) { dx[i] = srcRow[xoff + i]; df[i] = srcRow[foff + i]; }
  }
  __syncthreads();

  int w = tid >> 5, lane = tid & 31, half = lane >> 4, lo = lane & 15;
  int m_tile = w & 3, group = w >> 2;
  float t = temperature[hh];
  t = fminf(fmaxf(t, 0.1f), 5.0f);
  float invt = 1.0f / t;

  { // logits = x_mid @ Wslice^T : M=tok N=g K=d
    v8f l0 = v8f_zero(), l1 = v8f_zero();
    int tok = m_tile * 16 + lo;
    int g0 = group * 32 + lo;
    int g1 = group * 32 + 16 + lo;
    const __bf16* br0 = wsl + g0 * DH + half * 16;
    const __bf16* br1 = wsl + g1 * DH + half * 16;
#pragma unroll
    for (int ks = 0; ks < 2; ++ks) {
      int kb = ks * 32;
      v16bf a, b0, b1;
#pragma unroll
      for (int e = 0; e < 16; ++e) a[e] = lds_xm[tok * 64 + kb + a_kofs(e, half)];
#pragma unroll
      for (int e = 0; e < 16; ++e) b0[e] = br0[kb + e];
#pragma unroll
      for (int e = 0; e < 16; ++e) b1[e] = br1[kb + e];
      l0 = wmma_bf16(a, b0, l0);
      l1 = wmma_bf16(a, b1, l1);
    }
    float bs0 = bslice[g0], bs1 = bslice[g1];
#pragma unroll
    for (int v = 0; v < 8; ++v) {
      int row = m_tile * 16 + v + 8 * half;
      lds_lg[row * 65 + g0] = (l0[v] + bs0) * invt;
      lds_lg[row * 65 + g1] = (l1[v] + bs1) * invt;
    }
  }
  __syncthreads();

  // softmax per token row; write sw (bf16) transposed to LDS + g-major global
  if (tid < 64) {
    const float* row = lds_lg + tid * 65;
    float mx = -1e30f;
    for (int g = 0; g < GSL; ++g) mx = fmaxf(mx, row[g]);
    float s = 0.0f;
    for (int g = 0; g < GSL; ++g) s += __expf(row[g] - mx);
    float inv = 1.0f / s;
    __bf16* gdst = swg_all + (size_t)bh * GSL * NTOK + n0 + tid;
    for (int g = 0; g < GSL; ++g) {
      __bf16 hv = (__bf16)(__expf(row[g] - mx) * inv);
      lds_swT[g * 64 + tid] = hv;
      gdst[(size_t)g * NTOK] = hv;
    }
  }
  __syncthreads();

  if (tid < 64) { // slice_norm partial
    float s = 0.0f;
    for (int tk = 0; tk < 64; ++tk) s += (float)lds_swT[tid * 64 + tk];
    atomicAdd(&norm[bh * GSL + tid], s);
  }

  { // slice_token partial: D[g][d] = sum_tok swT[g][tok]*fx[tok][d]
    v8f s0 = v8f_zero(), s1 = v8f_zero();
    int gRow = m_tile * 16 + lo;
    int d0 = group * 32 + lo;
    int d1 = group * 32 + 16 + lo;
#pragma unroll
    for (int ks = 0; ks < 2; ++ks) {
      int kb = ks * 32;
      v16bf a, b0, b1;
#pragma unroll
      for (int e = 0; e < 16; ++e) a[e] = lds_swT[gRow * 64 + kb + a_kofs(e, half)];
#pragma unroll
      for (int e = 0; e < 16; ++e) b0[e] = lds_fx[(kb + half * 16 + e) * 64 + d0];
#pragma unroll
      for (int e = 0; e < 16; ++e) b1[e] = lds_fx[(kb + half * 16 + e) * 64 + d1];
      s0 = wmma_bf16(a, b0, s0);
      s1 = wmma_bf16(a, b1, s1);
    }
    float* dst = acc + (size_t)bh * GSL * DH;
#pragma unroll
    for (int v = 0; v < 8; ++v) {
      int g = m_tile * 16 + v + 8 * half;
      atomicAdd(&dst[g * DH + d0], s0[v]);
      atomicAdd(&dst[g * DH + d1], s1[v]);
    }
  }
}

// ---------------------------------------------------------------------------
// Kernel C: tiny 64x64 attention per (b,h) + fold Wout -> P[b,h][64][128] bf16
// ---------------------------------------------------------------------------
__global__ __launch_bounds__(256) void slice_attn(
    const float* __restrict__ acc, const float* __restrict__ norm,
    const float* __restrict__ Wq, const float* __restrict__ Wk,
    const float* __restrict__ Wv, const float* __restrict__ Wout,
    __bf16* __restrict__ P) {
  extern __shared__ char smem_raw[];
  float* st = (float*)smem_raw;   // slice_token, later attn
  float* qm = st + 4096;
  float* km = qm + 4096;          // later out_slice
  float* vm = km + 4096;

  int bh = blockIdx.x;
  int hh = bh & 7;
  int tid = threadIdx.x;

  for (int i = tid; i < 4096; i += 256) {
    int g = i >> 6;
    st[i] = acc[(size_t)bh * 4096 + i] / (norm[bh * GSL + g] + 1e-5f);
  }
  __syncthreads();
  for (int i = tid; i < 4096; i += 256) {
    int g = i >> 6, e = i & 63;
    float sq = 0.f, sk = 0.f, sv = 0.f;
    for (int d = 0; d < 64; ++d) {
      float xv = st[g * 64 + d];
      sq += xv * Wq[e * 64 + d];
      sk += xv * Wk[e * 64 + d];
      sv += xv * Wv[e * 64 + d];
    }
    qm[i] = sq; km[i] = sk; vm[i] = sv;
  }
  __syncthreads();
  for (int i = tid; i < 4096; i += 256) {  // attn logits -> st
    int g = i >> 6, f = i & 63;
    float s = 0.f;
    for (int e = 0; e < 64; ++e) s += qm[g * 64 + e] * km[f * 64 + e];
    st[i] = s * 0.125f;   // DIM_HEAD^-0.5
  }
  __syncthreads();
  if (tid < 64) {  // softmax rows of attn
    float mx = -1e30f;
    for (int f = 0; f < 64; ++f) mx = fmaxf(mx, st[tid * 64 + f]);
    float s = 0.f;
    for (int f = 0; f < 64; ++f) s += __expf(st[tid * 64 + f] - mx);
    float inv = 1.f / s;
    for (int f = 0; f < 64; ++f) st[tid * 64 + f] = __expf(st[tid * 64 + f] - mx) * inv;
  }
  __syncthreads();
  for (int i = tid; i < 4096; i += 256) {  // out_slice -> km
    int g = i >> 6, e = i & 63;
    float s = 0.f;
    for (int f = 0; f < 64; ++f) s += st[g * 64 + f] * vm[f * 64 + e];
    km[i] = s;
  }
  __syncthreads();
  for (int i = tid; i < 8192; i += 256) {  // P = out_slice @ Wout_head^T
    int g = i >> 7, o = i & 127;
    float s = 0.f;
    for (int e = 0; e < 64; ++e) s += km[g * 64 + e] * Wout[o * INNER + hh * 64 + e];
    P[(size_t)bh * 8192 + i] = (__bf16)s;
  }
}

// ---------------------------------------------------------------------------
// Kernel D: out[n][o] = sum_{h,g} sw[h][g][n] * P[h][g][o] + bout[o]
// GEMM per 64-token tile: M=64, N=128, K=512 (8 heads x 64 g)
// ---------------------------------------------------------------------------
__global__ __launch_bounds__(256) void out_kernel(
    const __bf16* __restrict__ swg_all, const __bf16* __restrict__ P,
    const float* __restrict__ bout, float* __restrict__ out) {
  extern __shared__ char smem_raw[];
  __bf16* lds_A = (__bf16*)smem_raw;     // [64 tok][64 g]
  __bf16* lds_B = lds_A + 64 * 64;       // [128 o][64 g]

  int blk  = blockIdx.x;
  int tile = blk & 1023;
  int b    = blk >> 10;
  int m0   = tile * 64;
  int tid  = threadIdx.x;
  int w = tid >> 5, lane = tid & 31, half = lane >> 4, lo = lane & 15;
  int m_tile = w & 3, group = w >> 2;
  int tok = m_tile * 16 + lo;
  int ob  = group * 64;

  v8f c0 = v8f_zero(), c1 = v8f_zero(), c2 = v8f_zero(), c3 = v8f_zero();

  for (int hh = 0; hh < HEADS; ++hh) {
    __syncthreads();
    int bh = b * HEADS + hh;
    { // stage A: sw tile, transpose g-major -> token-major
      int g = tid >> 2;
      int c = (tid & 3) << 4;
      const uint32_t* src =
          (const uint32_t*)(swg_all + ((size_t)bh * GSL + g) * NTOK + m0 + c);
#pragma unroll
      for (int i = 0; i < 8; ++i) {
        BF2 t2; t2.u = src[i];
        int tk = c + 2 * i;
        lds_A[tk * 64 + g] = t2.h[0];
        lds_A[(tk + 1) * 64 + g] = t2.h[1];
      }
    }
    { // stage B: P[g][o] -> lds_B[o][g] (K-contiguous for B fragments)
      const uint32_t* pb = (const uint32_t*)(P + (size_t)bh * 8192);
      for (int i = tid; i < 4096; i += 256) {
        int g = i >> 6, op = i & 63;
        BF2 t2; t2.u = pb[g * 64 + op];
        lds_B[(op * 2) * 64 + g] = t2.h[0];
        lds_B[(op * 2 + 1) * 64 + g] = t2.h[1];
      }
    }
    __syncthreads();
#pragma unroll
    for (int ks = 0; ks < 2; ++ks) {
      int kb = ks * 32;
      v16bf a, b0, b1, b2, b3;
#pragma unroll
      for (int e = 0; e < 16; ++e) a[e] = lds_A[tok * 64 + kb + a_kofs(e, half)];
#pragma unroll
      for (int e = 0; e < 16; ++e) {
        int kk = kb + half * 16 + e;
        b0[e] = lds_B[(ob + lo) * 64 + kk];
        b1[e] = lds_B[(ob + 16 + lo) * 64 + kk];
        b2[e] = lds_B[(ob + 32 + lo) * 64 + kk];
        b3[e] = lds_B[(ob + 48 + lo) * 64 + kk];
      }
      c0 = wmma_bf16(a, b0, c0);
      c1 = wmma_bf16(a, b1, c1);
      c2 = wmma_bf16(a, b2, c2);
      c3 = wmma_bf16(a, b3, c3);
    }
  }

  float bo0 = bout[ob + lo], bo1 = bout[ob + 16 + lo];
  float bo2 = bout[ob + 32 + lo], bo3 = bout[ob + 48 + lo];
#pragma unroll
  for (int v = 0; v < 8; ++v) {
    int row = m_tile * 16 + v + 8 * half;
    float* dst = out + (size_t)(b * NTOK + m0 + row) * CIN;
    dst[ob + lo]      = c0[v] + bo0;
    dst[ob + 16 + lo] = c1[v] + bo1;
    dst[ob + 32 + lo] = c2[v] + bo2;
    dst[ob + 48 + lo] = c3[v] + bo3;
  }
}

// ---------------------------------------------------------------------------
// Host launcher
// ---------------------------------------------------------------------------
extern "C" void kernel_launch(void* const* d_in, const int* in_sizes, int n_in,
                              void* d_out, int out_size, void* d_ws, size_t ws_size,
                              hipStream_t stream) {
  (void)in_sizes; (void)n_in; (void)out_size; (void)ws_size;
  const float* x    = (const float*)d_in[0];
  const float* Wfx  = (const float*)d_in[1];
  const float* bfx  = (const float*)d_in[2];
  const float* Wx   = (const float*)d_in[3];
  const float* bx   = (const float*)d_in[4];
  const float* Wsl  = (const float*)d_in[5];
  const float* bsl  = (const float*)d_in[6];
  const float* temp = (const float*)d_in[7];
  const float* Wq   = (const float*)d_in[8];
  const float* Wk   = (const float*)d_in[9];
  const float* Wv   = (const float*)d_in[10];
  const float* Wout = (const float*)d_in[11];
  const float* bout = (const float*)d_in[12];
  float* out = (float*)d_out;

  char* ws = (char*)d_ws;
  size_t off = 0;
  auto alloc = [&](size_t bytes) -> char* {
    char* p = ws + off;
    off += (bytes + 255) & ~(size_t)255;
    return p;
  };
  __bf16* wT   = (__bf16*)alloc((size_t)OC2 * KTOT * 2);            // 2.4 MB
  __bf16* wsl  = (__bf16*)alloc((size_t)GSL * DH * 2);              // 8 KB
  __bf16* xbf  = (__bf16*)alloc((size_t)BATCH * NTOK * CIN * 2);    // 32 MB
  __bf16* mid  = (__bf16*)alloc((size_t)BATCH * NTOK * OC2 * 2);    // 256 MB
  __bf16* swg  = (__bf16*)alloc((size_t)BATCH * HEADS * GSL * NTOK * 2); // 128 MB
  float*  norm = (float*)alloc((size_t)BATCH * HEADS * GSL * 4);
  float*  acc  = (float*)alloc((size_t)BATCH * HEADS * GSL * DH * 4);
  __bf16* P    = (__bf16*)alloc((size_t)BATCH * HEADS * GSL * CIN * 2);

  prep_weights<<<(OC2 * KTOT + 255) / 256, 256, 0, stream>>>(Wfx, Wx, wT);
  int nx = BATCH * NTOK * CIN;
  prep_x<<<(nx + 255) / 256, 256, 0, stream>>>(x, xbf, nx);
  prep_misc<<<(BATCH * HEADS * GSL * DH + 255) / 256, 256, 0, stream>>>(Wsl, wsl, norm, acc);

  // conv: B * 1024 pixel-tiles * 4 oc-slabs (256 oc each)
  conv_dual<<<BATCH * 1024 * 4, 256, 198 * 128 * 2, stream>>>(xbf, wT, bfx, bx, mid);
  // slices: B*H * 1024 token-tiles
  slice_kernel<<<BATCH * HEADS * 1024, 256,
                 (3 * 64 * 64) * 2 + 64 * 65 * 4, stream>>>(
      mid, wsl, bsl, temp, swg, norm, acc);
  slice_attn<<<BATCH * HEADS, 256, 4 * 4096 * 4, stream>>>(
      acc, norm, Wq, Wk, Wv, Wout, P);
  out_kernel<<<BATCH * 1024, 256, (64 * 64 + 128 * 64) * 2, stream>>>(
      swg, P, bout, out);
}
